// VecTurBlock_65034394796863
// MI455X (gfx1250) — matrix operand order repaired and checked by reading, not compiled, not run on hardware
//
#include <hip/hip_runtime.h>
#include <hip/hip_bf16.h>
#include <math.h>

// ---------------- problem constants (match reference) ----------------
constexpr int BB = 4;          // batch
constexpr int TT = 4096;       // tokens / tape slots
constexpr int DD = 1024;       // model dim
constexpr int KK = 8;          // heads
constexpr int TMAX = 4;
constexpr int HID2 = 8192;     // 4*2*D  (dq/dt hidden)
constexpr int HID4 = 4096;     // 4*D    (kappa hidden)
constexpr int PHN  = 8204;     // 4*(2D+3) per-head hidden
constexpr float EPS_F = 1e-6f;
constexpr float HALT_EPS = 1e-3f;
constexpr float TWO_PI_F = 6.283185307179586f;

typedef __attribute__((ext_vector_type(16))) __bf16 v16bf;
typedef __attribute__((ext_vector_type(8)))  float  v8f;

// ---------------- helpers ----------------
__device__ __forceinline__ unsigned short f2bf(float f) {      // RNE f32->bf16
  unsigned u = __float_as_uint(f);
  u += 0x7fffu + ((u >> 16) & 1u);
  return (unsigned short)(u >> 16);
}
__device__ __forceinline__ float bf2f(unsigned short h) {
  return __uint_as_float(((unsigned)h) << 16);
}
__device__ __forceinline__ float mod2pi(float x) {
  float r = fmodf(x, TWO_PI_F);
  return (r < 0.f) ? r + TWO_PI_F : r;
}
__device__ __forceinline__ float siluf(float x) { return x / (1.f + expf(-x)); }
__device__ __forceinline__ float clampf(float x, float lo, float hi) {
  return fminf(fmaxf(x, lo), hi);
}

// =====================================================================
// Kernel: tape = x @ m_t_w   (M=16384, N=1024, K=1024, fp32 via bf16x3)
// Block: 256 thr = 8 waves. Block tile 128(M) x 64(N); wave tile 16x64
// (4 N-subtiles -> 12 wmma per k-step per wave, amortizing A fragments).
// Double-buffered LDS pipeline: stage tile k+1 while issuing WMMAs on
// tile k; one barrier per iteration. global_prefetch on tile k+2.
// LDS layout is fragment-native: one 32B contiguous read per lane.
//   A row-major: pos(k) = k
//     -> lanes 0-15 read K0..15, lanes 16-31 read K16..31
//   B col-major: col*stride + k  -> lanes 0-15 K0..15, lanes 16-31 K16..31
// =====================================================================
__global__ __launch_bounds__(256) void k_tape_gemm(
    const float* __restrict__ x, const float* __restrict__ wmat,
    float* __restrict__ out) {
  __shared__ unsigned short Ah[2][128 * 48];   // hi parts, row stride 48 (96B)
  __shared__ unsigned short Al[2][128 * 48];   // lo parts
  __shared__ unsigned short Bh[2][64 * 48];
  __shared__ unsigned short Bl[2][64 * 48];

  const int tid  = threadIdx.x;
  const int wv   = tid >> 5;
  const int lane = tid & 31;
  const int l16  = lane & 15;
  const int half = lane >> 4;
  const int m0 = blockIdx.x * 128;
  const int n0 = blockIdx.y * 64;

  v8f acc[4];
#pragma unroll
  for (int t = 0; t < 4; ++t) acc[t] = (v8f){0.f, 0.f, 0.f, 0.f, 0.f, 0.f, 0.f, 0.f};

  const int lr = tid >> 3;              // 0..31
  const int lq = tid & 7;               // float4 index within a 32-wide row
  const int k4 = 4 * lq;
  const int kpos = k4; // contiguous x4

  auto stage = [&](int buf, int k0) {
    // ---- stage A tile: 128x32 f32 -> bf16 hi/lo, contiguous layout ----
#pragma unroll
    for (int p = 0; p < 4; ++p) {
      const int row = p * 32 + lr;
      const float4 v = *(const float4*)(x + (size_t)(m0 + row) * DD + k0 + k4);
      unsigned short hx = f2bf(v.x), hy = f2bf(v.y), hz = f2bf(v.z), hw = f2bf(v.w);
      unsigned short lx = f2bf(v.x - bf2f(hx)), ly = f2bf(v.y - bf2f(hy));
      unsigned short lz = f2bf(v.z - bf2f(hz)), lw = f2bf(v.w - bf2f(hw));
      uint2 ph; ph.x = (unsigned)hx | ((unsigned)hy << 16);
      ph.y = (unsigned)hz | ((unsigned)hw << 16);
      uint2 pl; pl.x = (unsigned)lx | ((unsigned)ly << 16);
      pl.y = (unsigned)lz | ((unsigned)lw << 16);
      *(uint2*)&Ah[buf][row * 48 + kpos] = ph;
      *(uint2*)&Al[buf][row * 48 + kpos] = pl;
    }
    // ---- stage B tile: 32(K) x 64(N), col-major in LDS ----
#pragma unroll
    for (int h2 = 0; h2 < 2; ++h2) {
      const float4 v =
          *(const float4*)(wmat + (size_t)(k0 + lr) * DD + n0 + 32 * h2 + k4);
      const float f[4] = {v.x, v.y, v.z, v.w};
#pragma unroll
      for (int i = 0; i < 4; ++i) {
        const int col = 32 * h2 + k4 + i;
        unsigned short hi = f2bf(f[i]);
        Bh[buf][col * 48 + lr] = hi;
        Bl[buf][col * 48 + lr] = f2bf(f[i] - bf2f(hi));
      }
    }
  };

  stage(0, 0);
  __syncthreads();

  constexpr int NK = DD / 32;           // 32 k-steps
  for (int kt = 0; kt < NK; ++kt) {
    const int cur = kt & 1;
    if (kt + 1 < NK) {
      stage(cur ^ 1, (kt + 1) * 32);
      if (kt + 2 < NK) {                // warm GL2 for the tile after next
        __builtin_prefetch(x + (size_t)(m0 + lr) * DD + (kt + 2) * 32 + k4, 0, 0);
        __builtin_prefetch(wmat + (size_t)((kt + 2) * 32 + lr) * DD + n0 + k4, 0, 0);
      }
    }
    const int arow = (wv * 16 + l16) * 48 + half * 16;
    const v16bf ah = *(const v16bf*)&Ah[cur][arow];
    const v16bf al = *(const v16bf*)&Al[cur][arow];
#pragma unroll
    for (int t = 0; t < 4; ++t) {
      const int bo = (16 * t + l16) * 48 + half * 16;
      const v16bf bh = *(const v16bf*)&Bh[cur][bo];
      const v16bf bl = *(const v16bf*)&Bl[cur][bo];
      // bf16x3 split-precision accumulate: hi*hi + hi*lo + lo*hi
      acc[t] = __builtin_amdgcn_wmma_f32_16x16x32_bf16(false, ah, false, bh,
                                                       (short)0, acc[t], false, false);
      acc[t] = __builtin_amdgcn_wmma_f32_16x16x32_bf16(false, ah, false, bl,
                                                       (short)0, acc[t], false, false);
      acc[t] = __builtin_amdgcn_wmma_f32_16x16x32_bf16(false, al, false, bh,
                                                       (short)0, acc[t], false, false);
    }
    __syncthreads();   // nxt staged for kt+1; cur free for restage at kt+1
  }

  // C/D layout: VGPR e -> M = e + 8*half, N = l16
#pragma unroll
  for (int t = 0; t < 4; ++t) {
    const int col = n0 + 16 * t + l16;
#pragma unroll
    for (int e = 0; e < 8; ++e) {
      const int row = m0 + wv * 16 + half * 8 + e;
      out[(size_t)row * DD + col] = acc[t][e];
    }
  }
}

// =====================================================================
// Setup kernels (bandwidth-bound, no matrix cores needed)
// =====================================================================
__global__ __launch_bounds__(256) void k_colsum(const float* __restrict__ x,
                                                float* __restrict__ xs) {
  const int idx = blockIdx.x * 256 + threadIdx.x;   // over B*D
  const int b = idx >> 10, d = idx & 1023;
  float a = 0.f;
  for (int t = 0; t < TT; ++t) a += x[((size_t)b * TT + t) * DD + d];
  xs[idx] = a;
}

__global__ __launch_bounds__(256) void k_qinit(const float* __restrict__ xs,
                                               const float* __restrict__ mqw,
                                               float* __restrict__ q) {
  const int idx = blockIdx.x * 256 + threadIdx.x;   // over B*D
  const int b = idx >> 10, d = idx & 1023;
  float a = 0.f;
  const float* xb = xs + b * DD;
  for (int i = 0; i < DD; ++i) a += xb[i] * mqw[(size_t)i * DD + d];
  q[idx] = a;
}

__global__ void k_proj_small(const float* __restrict__ xs,
                             const float* __restrict__ mth,
                             const float* __restrict__ mw,
                             float* __restrict__ theta, float* __restrict__ wst) {
  const int tid = threadIdx.x;   // 64 threads
  if (tid < BB * KK) {
    const int b = tid / KK, k = tid % KK;
    float a = 0.f; const float* xb = xs + b * DD;
    for (int i = 0; i < DD; ++i) a += xb[i] * mth[i * KK + k];
    theta[tid] = mod2pi(a);
  } else if (tid < 2 * BB * KK) {
    const int t2 = tid - BB * KK, b = t2 / KK, k = t2 % KK;
    float a = 0.f; const float* xb = xs + b * DD;
    for (int i = 0; i < DD; ++i) a += xb[i] * mw[i * KK + k];
    wst[t2] = a;
  }
}

__global__ __launch_bounds__(256) void k_kappa(const float* __restrict__ xs,
                                               const float* __restrict__ kw1,
                                               const float* __restrict__ kw2,
                                               float* __restrict__ kappa) {
  const int b = blockIdx.x, tid = threadIdx.x;
  __shared__ float red[256];
  float acc = 0.f;
  const float* xb = xs + b * DD;
  for (int j = tid; j < HID4; j += 256) {
    float p = 0.f;
    for (int i = 0; i < DD; ++i) p += xb[i] * kw1[(size_t)i * HID4 + j];
    acc += siluf(p) * kw2[j];
  }
  red[tid] = acc; __syncthreads();
  for (int s = 128; s > 0; s >>= 1) { if (tid < s) red[tid] += red[tid + s]; __syncthreads(); }
  if (tid == 0) {
    float v = red[0];
    kappa[b] = ((v > 20.f) ? v : log1pf(expf(v))) + EPS_F;   // softplus + eps
  }
}

__global__ void k_set_active(int* a) { if (threadIdx.x == 0) *a = 1; }

// =====================================================================
// Step kernels (sequential refinement; all freeze via `active`)
// =====================================================================
__global__ __launch_bounds__(256) void k_gather_st(
    const float* __restrict__ tape, const float* __restrict__ q,
    const float* __restrict__ q0, const float* __restrict__ theta,
    const float* __restrict__ wst, const float* __restrict__ kappa,
    float* __restrict__ s_t, float* __restrict__ g,
    int* __restrict__ nidx, int* __restrict__ npidx,
    float* __restrict__ wn, float* __restrict__ wnp,
    float* __restrict__ sin_s, float* __restrict__ cos_s,
    const int* __restrict__ active, int step) {
  if (!*active) return;
  const int b = blockIdx.x, tid = threadIdx.x;
  __shared__ float red[256];
  __shared__ int sh_n[KK], sh_np[KK];
  __shared__ float sh_wn[KK], sh_wnp[KK];

  float acc = 0.f;
  for (int d = tid; d < DD; d += 256) {
    const float df = q[b * DD + d] - q0[d];
    acc += df * df;
  }
  red[tid] = acc; __syncthreads();
  for (int s = 128; s > 0; s >>= 1) { if (tid < s) red[tid] += red[tid + s]; __syncthreads(); }
  if (tid == 0) {
    const float denom = fmaxf(red[0], EPS_F);
    g[b] = 1.f / (1.f + expf(kappa[b] * (float)step / denom));  // sigmoid(-k*step/denom)
  }
  if (tid < KK) {
    const float th = mod2pi(theta[b * KK + tid]);
    const float u = (float)TT * th / TWO_PI_F;
    const float nf = floorf(u);
    int n = (int)nf; n = max(0, min(n, TT - 1));
    const float s = u - nf;
    const int np = (n + 1) & (TT - 1);
    const float wv = wst[b * KK + tid];
    sh_n[tid] = n; sh_np[tid] = np;
    sh_wn[tid] = wv * (1.f - s); sh_wnp[tid] = wv * s;
    nidx[b * KK + tid] = n; npidx[b * KK + tid] = np;
    wn[b * KK + tid] = sh_wn[tid]; wnp[b * KK + tid] = sh_wnp[tid];
    sin_s[b * KK + tid] = sinf(th); cos_s[b * KK + tid] = cosf(th);
  }
  __syncthreads();
  for (int d = tid; d < DD; d += 256) {
    float s = 0.f;
    for (int k = 0; k < KK; ++k)
      s += tape[((size_t)b * TT + sh_n[k]) * DD + d] * sh_wn[k] +
           tape[((size_t)b * TT + sh_np[k]) * DD + d] * sh_wnp[k];
    s_t[b * DD + d] = s;
  }
}

// h[b,j] = silu( [s_t, qin] . W1[:, j] )    (W1: 2048 x 8192, row-major)
__global__ __launch_bounds__(256) void k_mlp_hidden(
    const float* __restrict__ s_t, const float* __restrict__ qin,
    const float* __restrict__ W1, float* __restrict__ h,
    const int* __restrict__ active) {
  if (!*active) return;
  const int idx = blockIdx.x * 256 + threadIdx.x;  // over B*HID2
  const int b = idx >> 13, j = idx & (HID2 - 1);
  const float* sb = s_t + b * DD;
  const float* qb = qin + b * DD;
  const float* w = W1 + j;
  float a = 0.f;
  for (int i = 0; i < DD; ++i) a += sb[i] * w[(size_t)i * HID2];
  for (int i = 0; i < DD; ++i) a += qb[i] * w[(size_t)(DD + i) * HID2];
  h[idx] = siluf(a);
}

// q_new = q + g * clip(h @ W2, -5, 5)
__global__ __launch_bounds__(256) void k_mlp_out_q(
    const float* __restrict__ h, const float* __restrict__ W2,
    const float* __restrict__ q, const float* __restrict__ g,
    float* __restrict__ q_new, const int* __restrict__ active) {
  if (!*active) return;
  const int idx = blockIdx.x * 256 + threadIdx.x;  // over B*D
  const int b = idx >> 10, d = idx & 1023;
  const float* hb = h + b * HID2;
  float a = 0.f;
  for (int j = 0; j < HID2; ++j) a += hb[j] * W2[(size_t)j * DD + d];
  q_new[idx] = q[idx] + g[b] * clampf(a, -5.f, 5.f);
}

// Abase[b,j] = s_t[b].W1[0:D, j] + q_new[b].W1[D:2D, j]  (shared across K heads)
__global__ __launch_bounds__(256) void k_ph_base(
    const float* __restrict__ s_t, const float* __restrict__ qn,
    const float* __restrict__ W1, float* __restrict__ Ab,
    const int* __restrict__ active) {
  if (!*active) return;
  const int b = blockIdx.x / 33;
  const int j = (blockIdx.x % 33) * 256 + threadIdx.x;
  if (j >= PHN) return;
  const float* sb = s_t + b * DD;
  const float* qb = qn + b * DD;
  const float* w = W1 + j;
  float a = 0.f;
  for (int i = 0; i < DD; ++i) a += sb[i] * w[(size_t)i * PHN];
  for (int i = 0; i < DD; ++i) a += qb[i] * w[(size_t)(DD + i) * PHN];
  Ab[b * PHN + j] = a;
}

// Per-head dtheta/dw and state update (both MLPs fused; one block per (b,k))
__global__ __launch_bounds__(256) void k_ph_out(
    const float* __restrict__ AbTh, const float* __restrict__ AbW,
    const float* __restrict__ W1th, const float* __restrict__ W2th,
    const float* __restrict__ W1w, const float* __restrict__ W2w,
    const float* __restrict__ sin_s, const float* __restrict__ cos_s,
    const float* __restrict__ g, float* __restrict__ theta,
    float* __restrict__ wst, const int* __restrict__ active) {
  if (!*active) return;
  const int bk = blockIdx.x, b = bk >> 3, tid = threadIdx.x;
  __shared__ float r0[256], r1[256];
  const float sv = sin_s[bk], cv = cos_s[bk], wv = wst[bk];
  const size_t rs = (size_t)2048 * PHN, rc = (size_t)2049 * PHN, rw = (size_t)2050 * PHN;
  float ath = 0.f, aw = 0.f;
  for (int j = tid; j < PHN; j += 256) {
    const float base_th = AbTh[b * PHN + j];
    const float base_w  = AbW[b * PHN + j];
    const float h1 = base_th + sv * W1th[rs + j] + cv * W1th[rc + j] + wv * W1th[rw + j];
    ath += siluf(h1) * W2th[j];
    const float h2 = base_w + sv * W1w[rs + j] + cv * W1w[rc + j] + wv * W1w[rw + j];
    aw += siluf(h2) * W2w[j];
  }
  r0[tid] = ath; r1[tid] = aw; __syncthreads();
  for (int s = 128; s > 0; s >>= 1) {
    if (tid < s) { r0[tid] += r0[tid + s]; r1[tid] += r1[tid + s]; }
    __syncthreads();
  }
  if (tid == 0) {
    const float dth = clampf(r0[0], -1.f, 1.f);
    const float dw  = clampf(r1[0], -2.f, 2.f);
    theta[bk] = mod2pi(theta[bk] + g[b] * dth);
    wst[bk]   = clampf(wv + g[b] * dw, -10.f, 10.f);
  }
}

// u_t = clip(h @ dt_w2), scatter-add into circular tape slots, clip touched slots.
// (Full-tape clip in reference is a no-op away from written slots.)
__global__ __launch_bounds__(256) void k_ut_tape(
    const float* __restrict__ h, const float* __restrict__ W2,
    float* __restrict__ tape, const float* __restrict__ g,
    const int* __restrict__ nidx, const int* __restrict__ npidx,
    const float* __restrict__ wn, const float* __restrict__ wnp,
    const int* __restrict__ active) {
  if (!*active) return;
  const int idx = blockIdx.x * 256 + threadIdx.x;  // over B*D
  const int b = idx >> 10, d = idx & 1023;
  const float* hb = h + b * HID2;
  float a = 0.f;
  for (int j = 0; j < HID2; ++j) a += hb[j] * W2[(size_t)j * DD + d];
  a = clampf(a, -5.f, 5.f);
  const float gb = g[b];
  // each thread owns (b,d): repeated slot indices across k serialize safely here
  for (int k = 0; k < KK; ++k) {
    tape[((size_t)b * TT + nidx[b * KK + k]) * DD + d]  += gb * wn[b * KK + k] * a;
    tape[((size_t)b * TT + npidx[b * KK + k]) * DD + d] += gb * wnp[b * KK + k] * a;
  }
  for (int k = 0; k < KK; ++k) {
    const size_t i1 = ((size_t)b * TT + nidx[b * KK + k]) * DD + d;
    const size_t i2 = ((size_t)b * TT + npidx[b * KK + k]) * DD + d;
    tape[i1] = clampf(tape[i1], -10.f, 10.f);
    tape[i2] = clampf(tape[i2], -10.f, 10.f);
  }
}

__global__ void k_halt(const float* __restrict__ g, int* __restrict__ active) {
  if (threadIdx.x == 0 && *active) {
    bool all_small = true;
    for (int b = 0; b < BB; ++b) all_small = all_small && (g[b] < HALT_EPS);
    if (all_small) *active = 0;
  }
}

// =====================================================================
// Host launcher
// =====================================================================
extern "C" void kernel_launch(void* const* d_in, const int* in_sizes, int n_in,
                              void* d_out, int out_size, void* d_ws, size_t ws_size,
                              hipStream_t stream) {
  const float* x     = (const float*)d_in[0];
  // d_in[1] = freqs_cis: unused by the reference
  const float* m_t_w = (const float*)d_in[2];
  const float* m_q_w = (const float*)d_in[3];
  const float* m_th  = (const float*)d_in[4];
  const float* m_w   = (const float*)d_in[5];
  const float* kw1   = (const float*)d_in[6];
  const float* kw2   = (const float*)d_in[7];
  const float* dq1   = (const float*)d_in[8];
  const float* dq2   = (const float*)d_in[9];
  const float* dt1   = (const float*)d_in[10];
  const float* dt2   = (const float*)d_in[11];
  const float* dth1  = (const float*)d_in[12];
  const float* dth2  = (const float*)d_in[13];
  const float* dw1   = (const float*)d_in[14];
  const float* dw2   = (const float*)d_in[15];
  const float* q0    = (const float*)d_in[16];

  float* tape = (float*)d_out;              // (B, T, D) — fully rewritten each call
  float* ws   = (float*)d_ws;

  // workspace layout (floats)
  float* xs    = ws + 0;          // 4096
  float* qA    = ws + 4096;       // 4096
  float* qB    = ws + 8192;       // 4096
  float* theta = ws + 12288;      // 32
  float* wst   = ws + 12320;      // 32
  float* kappa = ws + 12352;      // 4
  float* g     = ws + 12356;      // 4
  float* sin_s = ws + 12360;      // 32
  float* cos_s = ws + 12392;      // 32
  float* wn    = ws + 12424;      // 32
  float* wnp   = ws + 12456;      // 32
  float* s_t   = ws + 12488;      // 4096
  float* h1    = ws + 16584;      // 32768
  float* AbTh  = ws + 49352;      // 32816
  float* AbW   = ws + 82168;      // 32816
  int* nidx    = (int*)(ws + 114984);  // 32
  int* npidx   = (int*)(ws + 115016);  // 32
  int* active  = (int*)(ws + 115048);  // 1

  k_set_active<<<1, 32, 0, stream>>>(active);

  // setup: column sum + projections (sum over T commutes with the matmuls)
  k_colsum<<<(BB * DD) / 256, 256, 0, stream>>>(x, xs);
  k_qinit<<<(BB * DD) / 256, 256, 0, stream>>>(xs, m_q_w, qA);
  k_proj_small<<<1, 64, 0, stream>>>(xs, m_th, m_w, theta, wst);
  k_kappa<<<BB, 256, 0, stream>>>(xs, kw1, kw2, kappa);

  // big GEMM: tape = x @ m_t_w  (WMMA bf16x3 split for fp32-class accuracy)
  dim3 gg((BB * TT) / 128, DD / 64);
  k_tape_gemm<<<gg, 256, 0, stream>>>(x, m_t_w, tape);

  // sequential refinement steps (fixed launch sequence; `active` freezes state)
  for (int step = 0; step < TMAX; ++step) {
    float* qin  = (step & 1) ? qB : qA;
    float* qout = (step & 1) ? qA : qB;
    k_gather_st<<<BB, 256, 0, stream>>>(tape, qin, q0, theta, wst, kappa,
                                        s_t, g, nidx, npidx, wn, wnp,
                                        sin_s, cos_s, active, step);
    k_mlp_hidden<<<(BB * HID2) / 256, 256, 0, stream>>>(s_t, qin, dq1, h1, active);
    k_mlp_out_q<<<(BB * DD) / 256, 256, 0, stream>>>(h1, dq2, qin, g, qout, active);
    k_ph_base<<<BB * 33, 256, 0, stream>>>(s_t, qout, dth1, AbTh, active);
    k_ph_base<<<BB * 33, 256, 0, stream>>>(s_t, qout, dw1, AbW, active);
    k_ph_out<<<BB * KK, 256, 0, stream>>>(AbTh, AbW, dth1, dth2, dw1, dw2,
                                          sin_s, cos_s, g, theta, wst, active);
    k_mlp_hidden<<<(BB * HID2) / 256, 256, 0, stream>>>(s_t, qout, dt1, h1, active);
    k_ut_tape<<<(BB * DD) / 256, 256, 0, stream>>>(h1, dt2, tape, g, nidx, npidx,
                                                   wn, wnp, active);
    k_halt<<<1, 32, 0, stream>>>(g, active);
  }
}
